// Language_SelfAttention_64183991271745
// MI455X (gfx1250) — 16-bit and 8-bit matrix operand order repaired and checked by reading, compiled, not run on hardware
//
#include <hip/hip_runtime.h>

#define LSEQ 256
#define DMODEL 768
#define HEADS 12
#define DHEAD 64
#define ROWS_PER_T 4096
#define PLANE 3145728
#define SPIKE_PLANE 196608
#define WMAT 589824
#define MASKN 4194304

typedef _Float16 f16;
typedef __attribute__((ext_vector_type(16))) _Float16 v16h;
typedef __attribute__((ext_vector_type(8)))  _Float16 v8h;
typedef __attribute__((ext_vector_type(4)))  _Float16 v4h;
typedef __attribute__((ext_vector_type(8)))  float    v8f;
typedef __attribute__((ext_vector_type(8)))  int      v8i;
typedef __attribute__((ext_vector_type(4)))  unsigned int u32x4;
typedef __attribute__((ext_vector_type(4)))  int      i32x4;
typedef __attribute__((ext_vector_type(4)))  float    f32x4;
typedef __attribute__((ext_vector_type(8)))  short    s16x8;
typedef unsigned long long ull;

union V16H { v16h v; v8h h[2]; };
union V8I  { v8i  v; ull q[4]; u32x4 u[2]; };

#define HAS_ASYNC_LDS __has_builtin(__builtin_amdgcn_global_load_async_to_lds_b128)
typedef int avec __attribute__((vector_size(16)));
typedef __attribute__((address_space(1))) avec g_avec;
typedef __attribute__((address_space(3))) avec l_avec;

__device__ __forceinline__ void cp16(const void* g, void* l) {
#if HAS_ASYNC_LDS
  __builtin_amdgcn_global_load_async_to_lds_b128(
      (g_avec*)(ull)g, (l_avec*)(unsigned int)(ull)l, 0, 0);
#else
  *(u32x4*)l = *(const u32x4*)g;
#endif
}

__device__ __forceinline__ void wait_async_lds() {
#if HAS_ASYNC_LDS
#if __has_builtin(__builtin_amdgcn_s_wait_asynccnt)
  __builtin_amdgcn_s_wait_asynccnt(0);
#else
  asm volatile("s_wait_asynccnt 0" ::: "memory");
#endif
#endif
}

__global__ __launch_bounds__(256) void k_lif_x(const float* __restrict__ x,
                                               f16* __restrict__ xs) {
  const int p = (blockIdx.x * 256 + threadIdx.x) * 4;
  float v[4] = {0.f, 0.f, 0.f, 0.f};
  #pragma unroll
  for (int t = 0; t < 4; ++t) {
    const f32x4 xv = *(const f32x4*)(x + (size_t)t * PLANE + p);
    v4h sp;
    #pragma unroll
    for (int e = 0; e < 4; ++e) {
      v[e] = 0.5f * (v[e] + xv[e]);
      const float s = (v[e] >= 1.0f) ? 1.0f : 0.0f;
      v[e] *= (1.0f - s);
      sp[e] = (f16)s;
    }
    *(v4h*)(xs + (size_t)t * PLANE + p) = sp;
  }
}

__global__ __launch_bounds__(256) void k_wcvt(const float* __restrict__ s,
                                              f16* __restrict__ d) {
  const int i = (blockIdx.x * 256 + threadIdx.x) * 4;
  const f32x4 sv = *(const f32x4*)(s + i);
  v4h dv;
  #pragma unroll
  for (int e = 0; e < 4; ++e) dv[e] = (f16)sv[e];
  *(v4h*)(d + i) = dv;
}

__global__ __launch_bounds__(256) void k_mcvt(const int* __restrict__ m,
                                              unsigned char* __restrict__ mu8) {
  const int i = (blockIdx.x * 256 + threadIdx.x) * 4;
  const i32x4 mv = *(const i32x4*)(m + i);
  unsigned int wrd = 0;
  #pragma unroll
  for (int e = 0; e < 4; ++e) wrd |= (mv[e] ? 0xFFu : 0u) << (8 * e);
  *(unsigned int*)(mu8 + i) = wrd;
}

__global__ __launch_bounds__(256) void k_bnprep(const float* __restrict__ g,
                                                const float* __restrict__ be,
                                                const float* __restrict__ me,
                                                const float* __restrict__ va,
                                                float* __restrict__ inv,
                                                float* __restrict__ add) {
  const int j = blockIdx.x * 256 + threadIdx.x;
  if (j < DMODEL) {
    const float iv = g[j] / sqrtf(va[j] + 1e-5f);
    inv[j] = iv;
    add[j] = be[j] - me[j] * iv;
  }
}

template<int MODE>
__global__ __launch_bounds__(256) void k_gemm(const f16* __restrict__ A,
                                              const f16* __restrict__ Wt,
                                              const float* __restrict__ bninv,
                                              const float* __restrict__ bnadd,
                                              unsigned char* __restrict__ spk,
                                              float* __restrict__ outp,
                                              float vth) {
  __shared__ f16 Asm[2][4][16][64];
  __shared__ f16 Bsm[2][128][64];
  const int tid  = threadIdx.x;
  const int w    = tid >> 5, lane = tid & 31;
  const int lm   = lane & 15, lh = lane >> 4;
  const int rb   = blockIdx.x;
  const int rbase = rb * 16;
  const int n0   = blockIdx.y * 128;

  v8f acc[4] = {};

  const int a_row = tid >> 2;
  const int a_t = a_row >> 4, a_m = a_row & 15, a_c = (tid & 3) * 8;
  const int b_row = tid >> 1, b_c = (tid & 1) * 32;
  const f16* agb = A + (size_t)(a_t * ROWS_PER_T + rbase + a_m) * DMODEL + a_c;
  const f16* bgb = Wt + (size_t)(n0 + b_row) * DMODEL + b_c;

  {
    cp16(agb,      &Asm[0][a_t][a_m][a_c]);
    cp16(agb + 32, &Asm[0][a_t][a_m][a_c + 32]);
    cp16(bgb,      &Bsm[0][b_row][b_c]);
    cp16(bgb + 8,  &Bsm[0][b_row][b_c + 8]);
    cp16(bgb + 16, &Bsm[0][b_row][b_c + 16]);
    cp16(bgb + 24, &Bsm[0][b_row][b_c + 24]);
  }

  int buf = 0;
  for (int k0 = 0; k0 < DMODEL; k0 += 64) {
    wait_async_lds();
    __syncthreads();
    if (k0 + 64 < DMODEL) {
      const f16* ag = agb + (k0 + 64);
      const f16* bg = bgb + (k0 + 64);
      const int nb = buf ^ 1;
      cp16(ag,      &Asm[nb][a_t][a_m][a_c]);
      cp16(ag + 32, &Asm[nb][a_t][a_m][a_c + 32]);
      cp16(bg,      &Bsm[nb][b_row][b_c]);
      cp16(bg + 8,  &Bsm[nb][b_row][b_c + 8]);
      cp16(bg + 16, &Bsm[nb][b_row][b_c + 16]);
      cp16(bg + 24, &Bsm[nb][b_row][b_c + 24]);
      __builtin_prefetch(ag + 64, 0, 1);
      __builtin_prefetch(bg + 64, 0, 1);
    }
    #pragma unroll
    for (int ks = 0; ks < 64; ks += 32) {
      V16H bf;
      const f16* bp = &Bsm[buf][w * 16 + lm][ks + lh * 16];
      bf.h[0] = *(const v8h*)bp;
      bf.h[1] = *(const v8h*)(bp + 8);
      #pragma unroll
      for (int t = 0; t < 4; ++t) {
        V16H af;
        const f16* ap = &Asm[buf][t][lm][ks + lh * 16];
        af.h[0] = *(const v8h*)ap;
        af.h[1] = *(const v8h*)(ap + 8);
        acc[t] = __builtin_amdgcn_wmma_f32_16x16x32_f16(false, af.v, false, bf.v,
                                                        (short)0, acc[t], false, false);
      }
    }
    buf ^= 1;
  }

  const int n = n0 + w * 16 + lm;
  const float inv = bninv[n], add = bnadd[n];
  const int bidx  = rbase >> 8;
  const int lbase = (rbase & 255) + lh * 8;

  if (MODE == 0) {
    float vmem[8];
    #pragma unroll
    for (int g = 0; g < 8; ++g) vmem[g] = 0.f;
    #pragma unroll
    for (int t = 0; t < 4; ++t) {
      ull pk = 0;
      #pragma unroll
      for (int g = 0; g < 8; ++g) {
        const float y = acc[t][g] * inv + add;
        vmem[g] = 0.5f * (vmem[g] + y);
        const float s = (vmem[g] >= vth) ? 1.0f : 0.0f;
        vmem[g] *= (1.0f - s);
        pk |= (ull)(s != 0.0f ? 1u : 0u) << (8 * g);
      }
      *(ull*)(spk + (size_t)(t * 16 + bidx) * SPIKE_PLANE + (size_t)n * LSEQ + lbase) = pk;
    }
  } else {
    #pragma unroll
    for (int t = 0; t < 4; ++t) {
      f32x4 o0, o1;
      #pragma unroll
      for (int g = 0; g < 4; ++g) o0[g] = acc[t][g] * inv + add;
      #pragma unroll
      for (int g = 0; g < 4; ++g) o1[g] = acc[t][g + 4] * inv + add;
      float* op = outp + (size_t)(t * 16 + bidx) * SPIKE_PLANE + (size_t)n * LSEQ + lbase;
      *(f32x4*)op       = o0;
      *(f32x4*)(op + 4) = o1;
    }
  }
}

__global__ __launch_bounds__(256) void k_attn(const unsigned char* __restrict__ q,
                                              const unsigned char* __restrict__ k,
                                              const unsigned char* __restrict__ v,
                                              const unsigned char* __restrict__ mu8,
                                              short* __restrict__ o) {
  __shared__ unsigned char attn_lds[128 * 256];
  __shared__ unsigned char vt_lds[64 * 256];
  const int tbh = blockIdx.x;
  const int tb = tbh / HEADS, h = tbh % HEADS;
  const int tid = threadIdx.x;
  const int w = tid >> 5, lane = tid & 31;
  const int lm = lane & 15, lh = lane >> 4;

  const size_t sbase = (size_t)tb * SPIKE_PLANE + h * DHEAD;
  const unsigned char* qb = q + sbase;
  const unsigned char* kb = k + sbase;
  const unsigned char* vb = v + sbase;
  const unsigned char* mb8 = mu8 + (size_t)tb * (LSEQ * LSEQ);

  {
    const int j0 = (tid & 63) * 4, dc0 = (tid >> 6) * 16;
    union { u32x4 v4; unsigned char b[16]; } r[4];
    #pragma unroll
    for (int i = 0; i < 4; ++i)
      r[i].v4 = *(const u32x4*)(vb + (size_t)(j0 + i) * DMODEL + dc0);
    #pragma unroll
    for (int c = 0; c < 16; ++c) {
      const unsigned int wrd = (unsigned)r[0].b[c] | ((unsigned)r[1].b[c] << 8) |
                               ((unsigned)r[2].b[c] << 16) | ((unsigned)r[3].b[c] << 24);
      *(unsigned int*)&vt_lds[(dc0 + c) * 256 + j0] = wrd;
    }
  }

  for (int pass = 0; pass < 2; ++pass) {
    __syncthreads();
    {
      const int i0g = pass * 128 + w * 16;
      const int iloc = w * 16 + lm;
      V8I qf;
      const unsigned char* qr = qb + (size_t)(i0g + lm) * DMODEL + lh * 8;
      qf.q[0] = *(const ull*)qr;
      qf.q[1] = *(const ull*)(qr + 16);
      qf.q[2] = *(const ull*)(qr + 32);
      qf.q[3] = *(const ull*)(qr + 48);
      const unsigned char* mrow = mb8 + (size_t)(pass * 128 + iloc) * LSEQ + lh * 8;
      unsigned char* arow = &attn_lds[iloc * 256 + lh * 8];

      V8I kf;
      {
        const unsigned char* kr = kb + (size_t)lm * DMODEL + lh * 8;
        kf.q[0] = *(const ull*)kr;
        kf.q[1] = *(const ull*)(kr + 16);
        kf.q[2] = *(const ull*)(kr + 32);
        kf.q[3] = *(const ull*)(kr + 48);
      }
      ull mw = *(const ull*)mrow;

      for (int jb = 0; jb < 16; ++jb) {
        V8I kf_n = {};
        ull mw_n = 0;
        if (jb < 15) {
          const unsigned char* kr = kb + (size_t)((jb + 1) * 16 + lm) * DMODEL + lh * 8;
          kf_n.q[0] = *(const ull*)kr;
          kf_n.q[1] = *(const ull*)(kr + 16);
          kf_n.q[2] = *(const ull*)(kr + 32);
          kf_n.q[3] = *(const ull*)(kr + 48);
          mw_n = *(const ull*)(mrow + (jb + 1) * 16);
        }
        v8i c = {};
        c = __builtin_amdgcn_wmma_i32_16x16x64_iu8(false, kf.v, false, qf.v,
                                                   c, false, false);
        const unsigned int lo = ((unsigned)c[0] & 255) | (((unsigned)c[1] & 255) << 8) |
                                (((unsigned)c[2] & 255) << 16) | ((unsigned)c[3] << 24);
        const unsigned int hi = ((unsigned)c[4] & 255) | (((unsigned)c[5] & 255) << 8) |
                                (((unsigned)c[6] & 255) << 16) | ((unsigned)c[7] << 24);
        const ull pk = ((((ull)hi << 32) | lo)) & mw;
        *(ull*)(arow + jb * 16) = pk;
        kf = kf_n;
        mw = mw_n;
      }
    }
    __syncthreads();
    {
      v8i oacc[4] = {};
      for (int kc = 0; kc < 4; ++kc) {
        V8I bf;
        const unsigned char* ar = &attn_lds[(w * 16 + lm) * 256 + kc * 64 + lh * 8];
        bf.q[0] = *(const ull*)ar;
        bf.q[1] = *(const ull*)(ar + 16);
        bf.q[2] = *(const ull*)(ar + 32);
        bf.q[3] = *(const ull*)(ar + 48);
        #pragma unroll
        for (int c0 = 0; c0 < 4; ++c0) {
          V8I vf;
          const unsigned char* vr = &vt_lds[(c0 * 16 + lm) * 256 + kc * 64 + lh * 8];
          vf.q[0] = *(const ull*)vr;
          vf.q[1] = *(const ull*)(vr + 16);
          vf.q[2] = *(const ull*)(vr + 32);
          vf.q[3] = *(const ull*)(vr + 48);
          oacc[c0] = __builtin_amdgcn_wmma_i32_16x16x64_iu8(false, vf.v, false, bf.v,
                                                            oacc[c0], false, false);
        }
      }
      const int row = pass * 128 + w * 16 + lm;
      short* op = o + ((size_t)tb * LSEQ + row) * DMODEL + h * DHEAD;
      #pragma unroll
      for (int c0 = 0; c0 < 4; ++c0) {
        s16x8 pkt;
        #pragma unroll
        for (int g = 0; g < 8; ++g) pkt[g] = (short)oacc[c0][g];
        *(s16x8*)(op + c0 * 16 + lh * 8) = pkt;
      }
    }
  }
}

__global__ __launch_bounds__(256) void k_lif_o(const short* __restrict__ o,
                                               f16* __restrict__ os) {
  const int p = (blockIdx.x * 256 + threadIdx.x) * 8;
  float v[8] = {0.f, 0.f, 0.f, 0.f, 0.f, 0.f, 0.f, 0.f};
  #pragma unroll
  for (int t = 0; t < 4; ++t) {
    const s16x8 ov = *(const s16x8*)(o + (size_t)t * PLANE + p);
    v8h sp;
    #pragma unroll
    for (int e = 0; e < 8; ++e) {
      const float y = (float)ov[e] * 0.125f;
      v[e] = 0.5f * (v[e] + y);
      const float s = (v[e] >= 0.5f) ? 1.0f : 0.0f;
      v[e] *= (1.0f - s);
      sp[e] = (f16)s;
    }
    *(v8h*)(os + (size_t)t * PLANE + p) = sp;
  }
}

extern "C" void kernel_launch(void* const* d_in, const int* in_sizes, int n_in,
                              void* d_out, int out_size, void* d_ws, size_t ws_size,
                              hipStream_t stream) {
  (void)in_sizes; (void)n_in; (void)out_size; (void)ws_size;
  const float* x     = (const float*)d_in[0];
  const int*   amask = (const int*)d_in[1];
  const float* W[4]   = { (const float*)d_in[2],  (const float*)d_in[7],
                          (const float*)d_in[12], (const float*)d_in[17] };
  const float* BNg[4] = { (const float*)d_in[3],  (const float*)d_in[8],
                          (const float*)d_in[13], (const float*)d_in[18] };
  const float* BNb[4] = { (const float*)d_in[4],  (const float*)d_in[9],
                          (const float*)d_in[14], (const float*)d_in[19] };
  const float* BNm[4] = { (const float*)d_in[5],  (const float*)d_in[10],
                          (const float*)d_in[15], (const float*)d_in[20] };
  const float* BNv[4] = { (const float*)d_in[6],  (const float*)d_in[11],
                          (const float*)d_in[16], (const float*)d_in[21] };

  char* ws = (char*)d_ws;
  size_t off = 0;
  f16* xs = (f16*)(ws + off);            off += (size_t)4 * PLANE * 2;
  f16* w16 = (f16*)(ws + off);           off += (size_t)4 * WMAT * 2;
  float* bninv = (float*)(ws + off);     off += (size_t)4 * DMODEL * 4;
  float* bnadd = (float*)(ws + off);     off += (size_t)4 * DMODEL * 4;
  unsigned char* qs = (unsigned char*)(ws + off); off += (size_t)4 * PLANE;
  unsigned char* kk = (unsigned char*)(ws + off); off += (size_t)4 * PLANE;
  unsigned char* vs = (unsigned char*)(ws + off); off += (size_t)4 * PLANE;
  unsigned char* mu8 = (unsigned char*)(ws + off); off += MASKN;
  short* ot = (short*)(ws + off);        off += (size_t)4 * PLANE * 2;
  f16* os = (f16*)(ws + off);            off += (size_t)4 * PLANE * 2;

  k_lif_x<<<PLANE / 1024, 256, 0, stream>>>(x, xs);
  k_mcvt<<<MASKN / 1024, 256, 0, stream>>>(amask, mu8);
  for (int i = 0; i < 4; ++i)
    k_wcvt<<<WMAT / 1024, 256, 0, stream>>>(W[i], w16 + (size_t)i * WMAT);
  for (int i = 0; i < 4; ++i)
    k_bnprep<<<3, 256, 0, stream>>>(BNg[i], BNb[i], BNm[i], BNv[i],
                                    bninv + i * DMODEL, bnadd + i * DMODEL);

  dim3 gg(256, 6);
  k_gemm<0><<<gg, 256, 0, stream>>>(xs, w16 + (size_t)0 * WMAT, bninv + 0 * DMODEL,
                                    bnadd + 0 * DMODEL, qs, nullptr, 1.0f);
  k_gemm<0><<<gg, 256, 0, stream>>>(xs, w16 + (size_t)1 * WMAT, bninv + 1 * DMODEL,
                                    bnadd + 1 * DMODEL, kk, nullptr, 1.0f);
  k_gemm<0><<<gg, 256, 0, stream>>>(xs, w16 + (size_t)2 * WMAT, bninv + 2 * DMODEL,
                                    bnadd + 2 * DMODEL, vs, nullptr, 1.0f);

  k_attn<<<64 * HEADS, 256, 0, stream>>>(qs, kk, vs, mu8, ot);
  k_lif_o<<<PLANE / 2048, 256, 0, stream>>>(ot, os);

  k_gemm<1><<<gg, 256, 0, stream>>>(os, w16 + (size_t)3 * WMAT, bninv + 3 * DMODEL,
                                    bnadd + 3 * DMODEL, nullptr, (float*)d_out, 0.f);
}
